// Scalar3DHMM_21199958573840
// MI455X (gfx1250) — matrix operand order repaired and checked by reading, not compiled, not run on hardware
//
#include <hip/hip_runtime.h>

// Scalar3DHMM forward on gfx1250.
// Strategy: logsumexp recurrence rewritten as max-rescaled GEMM so the inner
// contraction runs on v_wmma_f32_16x16x32_f16. One persistent workgroup
// (32 waves) iterates all T steps with alpha/X in LDS; P=exp(logT) is f16 in
// global memory (512KB, L2-resident across the 39 reuses).

#define XS   500      // real states
#define SP   512      // padded states (multiple of 32 for WMMA K, 16 for N)
#define NB   32       // batch (stories)
#define NTT  40       // time steps
#define SL   20       // sentence length
#define NTOK 10000    // token vocabulary

typedef __attribute__((ext_vector_type(16))) _Float16 v16h;
typedef __attribute__((ext_vector_type(8)))  _Float16 v8h;
typedef __attribute__((ext_vector_type(8)))  float    v8f;

// ---------------------------------------------------------------------------
// Kernel 1: emission log-probs  em[b,t,s] = sum_l logE[s, tok[b,t,l]]
// One block per (b,t); 512 threads over padded state dim. The 20MB emission
// table becomes L2-resident, so the strided column gathers are L2 hits.
// ---------------------------------------------------------------------------
__global__ __launch_bounds__(SP) void hmm_emis(const float* __restrict__ logE,
                                               const int*   __restrict__ stories,
                                               float*       __restrict__ emis) {
  const int bt = blockIdx.x;          // b*NTT + t
  const int s  = threadIdx.x;         // 0..511
  const int* tok = stories + bt * SL;
  float sum = 0.0f;
  if (s < XS) {
#pragma unroll
    for (int l = 0; l < SL; ++l)
      sum += logE[(size_t)s * NTOK + tok[l]];
  }
  emis[(size_t)bt * SP + s] = sum;    // padded lanes get 0 (never consumed)
}

// ---------------------------------------------------------------------------
// Kernel 2: W[n,k] = (half) exp(logT[n,k]), zero-padded to 512x512.
// The -1e9 masked logits underflow to exactly 0 (correct: P(i->j)=0).
// ---------------------------------------------------------------------------
__global__ __launch_bounds__(256) void hmm_wexp(const float* __restrict__ logT,
                                                _Float16*    __restrict__ W) {
  const int idx = blockIdx.x * 256 + threadIdx.x;   // over SP*SP
  const int n = idx >> 9;
  const int k = idx & (SP - 1);
  float v = 0.0f;
  if (n < XS && k < XS) v = __expf(logT[(size_t)n * XS + k]);
  W[idx] = (_Float16)v;
}

// ---------------------------------------------------------------------------
// Kernel 3: persistent single-workgroup forward recurrence.
// 32 waves; wave w owns destination-state tile n0 = 16*w.
// Per step: (1) per-batch max + X = exp(alpha - m) into LDS (f16),
//           (2) Y[32,512] = X * W^T via WMMA, alpha' = em + m + log(Y).
// ---------------------------------------------------------------------------
__global__ __launch_bounds__(1024) void hmm_forward(const float*    __restrict__ priors,
                                                    const float*    __restrict__ emis,
                                                    const _Float16* __restrict__ W,
                                                    float*          __restrict__ out) {
  __shared__ float    alpha[NB][SP + 1];   // +1 f32 pad: break row-bank aliasing
  __shared__ _Float16 Xs[NB][SP + 8];      // +8 halfs (16B) pad: ds_b128 stays aligned,
                                           // rows hit rotated banks
  __shared__ float    mrow[NB];

  const int tid  = threadIdx.x;
  const int wave = tid >> 5;
  const int lane = tid & 31;
  const int half = lane >> 4;   // lane half per WMMA register layouts
  const int lr   = lane & 15;
  const int n0   = wave * 16;   // this wave's N tile

  // ---- t = 0 : alpha0 = em[:,0] + priors --------------------------------
  for (int idx = tid; idx < NB * SP; idx += 1024) {
    const int b = idx >> 9;
    const int s = idx & (SP - 1);
    float a = -1e30f;
    if (s < XS) {
      a = emis[(size_t)(b * NTT) * SP + s] + priors[s];
      out[(size_t)b * XS + s] = a;                 // out[0][b][s]
    }
    alpha[b][s] = a;
  }
  __syncthreads();

  for (int t = 1; t < NTT; ++t) {
    // ---- phase 1: wave w handles batch b=w: rowmax + rescaled exp -------
    {
      const int b = wave;
      float lm = -1e30f;
      for (int k = lane; k < XS; k += 32) lm = fmaxf(lm, alpha[b][k]);
#pragma unroll
      for (int off = 16; off > 0; off >>= 1)
        lm = fmaxf(lm, __shfl_xor(lm, off, 32));
      if (lane == 0) mrow[b] = lm;
      for (int k = lane; k < SP; k += 32) {
        const float xv = (k < XS) ? __expf(alpha[b][k] - lm) : 0.0f;
        Xs[b][k] = (_Float16)xv;
      }
    }
    __syncthreads();

    // ---- phase 2: Y = X * W^T, one 16-wide N tile per wave --------------
    for (int mt = 0; mt < 2; ++mt) {
      v8f acc = {};
      const int am = mt * 16 + lr;        // A row (batch) for this lane
#pragma unroll 4
      for (int kb = 0; kb < SP; kb += 32) {
        // A tile 16x32 f16: lanes 0-15 carry K kb..kb+15,
        // lanes 16-31 carry K kb+16..kb+31 (matches B)
        const int ka = kb + half * 16;
        const v8h lo = *(const v8h*)&Xs[am][ka];
        const v8h hi = *(const v8h*)&Xs[am][ka + 8];
        v16h av;
#pragma unroll
        for (int i = 0; i < 8; ++i) { av[i] = lo[i]; av[i + 8] = hi[i]; }
        // B tile 32x16 f16: lane = column n, V0..7 hold K kb..kb+15 (half 0)
        // or kb+16..kb+31 (half 1). W row-major [n][k] -> contiguous 32B.
        const v16h bv =
            *(const v16h*)(W + (size_t)(n0 + lr) * SP + kb + half * 16);
        acc = __builtin_amdgcn_wmma_f32_16x16x32_f16(
            false, av, false, bv, (short)0, acc, false, false);
      }
      // D layout: VGPR r -> M = mt*16 + 8*half + r, N = n0 + lr
      const int n = n0 + lr;
#pragma unroll
      for (int r = 0; r < 8; ++r) {
        const int m = mt * 16 + half * 8 + r;
        if (n < XS) {
          const float val = emis[(size_t)(m * NTT + t) * SP + n] + mrow[m] +
                            __logf(acc[r]);
          out[((size_t)t * NB + m) * XS + n] = val;
          alpha[m][n] = val;
        }
      }
    }
    __syncthreads();
  }
}

// ---------------------------------------------------------------------------
extern "C" void kernel_launch(void* const* d_in, const int* in_sizes, int n_in,
                              void* d_out, int out_size, void* d_ws, size_t ws_size,
                              hipStream_t stream) {
  (void)in_sizes; (void)n_in; (void)out_size; (void)ws_size;

  const float* priors  = (const float*)d_in[0];   // [500]
  const float* logT    = (const float*)d_in[1];   // [500,500]
  const float* logE    = (const float*)d_in[2];   // [500,10000]
  const int*   stories = (const int*)d_in[3];     // [32,40,20]
  // d_in[4] = story_length (compile-time constant NTT here)

  float* out = (float*)d_out;                     // [40,32,500]

  // workspace: em[B][T][SP] f32 (2.62MB) then W[SP][SP] f16 (0.5MB)
  float*    emis = (float*)d_ws;
  _Float16* W    = (_Float16*)((char*)d_ws + (size_t)NB * NTT * SP * sizeof(float));

  hmm_emis<<<NB * NTT, SP, 0, stream>>>(logE, stories, emis);
  hmm_wexp<<<(SP * SP) / 256, 256, 0, stream>>>(logT, W);
  hmm_forward<<<1, 1024, 0, stream>>>(priors, emis, W, out);
}
